// PersistentControlLayer_83159156785948
// MI455X (gfx1250) — matrix operand order repaired and checked by reading, not compiled, not run on hardware
//
#include <hip/hip_runtime.h>

// ---------------------------------------------------------------------------
// CDNA5 (gfx1250) fused PersistentControlLayer
// wave32, v_wmma_f32_16x16x32_bf16 for all GEMMs, f32 accumulation.
// ---------------------------------------------------------------------------

typedef __attribute__((ext_vector_type(16))) __bf16 v16bf;
typedef __attribute__((ext_vector_type(8)))  float  v8f;
typedef __attribute__((ext_vector_type(4)))  unsigned int v4u;

union FragU { v4u q[2]; v16bf v; };

__device__ __forceinline__ v8f wmma_bf16(v8f c, v16bf a, v16bf b) {
  return __builtin_amdgcn_wmma_f32_16x16x32_bf16(false, a, false, b, (short)0, c, false, false);
}
__device__ __forceinline__ v8f zero8() { v8f z = {0.f,0.f,0.f,0.f,0.f,0.f,0.f,0.f}; return z; }

#define Hh     2048
#define NTOK   32768          // 8*4096
#define MT     64             // tokens per workgroup
#define EPSF   1e-6f
#define ALPHAF 0.35f

// workspace layout (bytes)
#define WS_GATE 0             // gateT  bf16 [64][64]   (row r, contiguous k)   8192 B
#define WS_SN   8192          // scratch_norm bf16 [64][128]                    16384 B
#define WS_WU   24576         // Wu bf16 [2048][64] row-major                   262144 B

// LDS layout (bytes) -- 65536 total
#define L_ZLR   0             // f32 [64][192]: z cols 0..127, lr cols 128..191 (49152 B)
#define L_STAGE 49152         // 16384 B multipurpose (A/W chunks, z_norm, mp, x)
#define L_TOTAL 65536

extern "C" __global__ __launch_bounds__(256)
void pcl_main(const float* __restrict__ hid, const float* __restrict__ Wr,
              const float* __restrict__ Wd, const float* __restrict__ thr,
              const float* __restrict__ cmean, const float* __restrict__ cstd,
              const unsigned char* __restrict__ ws, float* __restrict__ out)
{
  __shared__ __align__(16) unsigned char sm[L_TOTAL];
  float*  zlr   = (float*)(sm + L_ZLR);            // [64][192] f32
  __bf16* stage = (__bf16*)(sm + L_STAGE);         // 8192 bf16 elements
  float*  red   = (float*)(sm + L_STAGE + 8192);   // 192 f32 (stats scratch, B2 only)
  float*  nrm   = (float*)(sm + L_STAGE);          // 64 f32 (||delta||^2, Phase C only)

  const int tid  = threadIdx.x;
  const int lane = tid & 31;
  const int wv   = tid >> 5;
  const int g    = lane >> 4;     // half-wave group
  const int l16  = lane & 15;
  const int T0   = blockIdx.x * MT;
  float* stats = out + (size_t)NTOK * Hh;
  const float invN = 1.0f / (float)NTOK;

  // ============ Phase A: z = h@Wr^T (64x128), lr = h@Wd^T (64x64) ============
  {
    const int mtile = wv >> 1;                 // wave-fixed M tile
    v8f acc[6];
    #pragma unroll
    for (int i = 0; i < 6; ++i) acc[i] = zero8();
    __bf16* smA = stage;                       // 64x32 bf16 (64B rows)
    __bf16* smW = stage + 2048;                // 192x32 bf16 (64B rows)

    for (int ks = 0; ks < Hh / 32; ++ks) {
      const int h0 = ks * 32;
      float ra[8];
      #pragma unroll
      for (int e = 0; e < 8; ++e) {
        int idx = tid + e * 256, r = idx >> 5, c = idx & 31;
        ra[e] = hid[(size_t)(T0 + r) * Hh + h0 + c];
      }
      float rw[24];
      #pragma unroll
      for (int e = 0; e < 24; ++e) {
        int idx = tid + e * 256, r = idx >> 5, c = idx & 31;
        rw[e] = (r < 128) ? Wr[(size_t)r * Hh + h0 + c]
                          : Wd[(size_t)(r - 128) * Hh + h0 + c];
      }
      __syncthreads();   // previous iteration's fragments consumed
      #pragma unroll
      for (int e = 0; e < 8; ++e)  smA[tid + e * 256] = (__bf16)ra[e];
      #pragma unroll
      for (int e = 0; e < 24; ++e) smW[tid + e * 256] = (__bf16)rw[e];
      __syncthreads();

      const unsigned char* abase = (const unsigned char*)smA + (mtile * 16 + l16) * 64;
      FragU ua;
      ua.q[0] = *(const v4u*)(abase + g * 32);
      ua.q[1] = *(const v4u*)(abase + g * 32 + 16);
      #pragma unroll
      for (int i = 0; i < 6; ++i) {
        const int ntile = (wv & 1) * 6 + i;
        const unsigned char* bbase = (const unsigned char*)smW + (ntile * 16 + l16) * 64;
        FragU ub;
        ub.q[0] = *(const v4u*)(bbase + g * 32);
        ub.q[1] = *(const v4u*)(bbase + g * 32 + 16);
        acc[i] = wmma_bf16(acc[i], ua.v, ub.v);
      }
    }
    __syncthreads();
    #pragma unroll
    for (int i = 0; i < 6; ++i) {
      const int col = ((wv & 1) * 6 + i) * 16 + l16;
      #pragma unroll
      for (int j = 0; j < 8; ++j)
        zlr[(mtile * 16 + j + g * 8) * 192 + col] = acc[i][j];
    }
  }
  __syncthreads();

  // ============ Phase B1: z_norm (bf16) into stage [64][128] ============
  if (tid < 64) {
    const int t = tid;
    float ss = 0.f;
    for (int r = 0; r < 128; ++r) { float v = zlr[t * 192 + r]; ss += v * v; }
    const float inv = 1.0f / fmaxf(sqrtf(ss), EPSF);
    for (int r = 0; r < 128; ++r) stage[t * 128 + r] = (__bf16)(zlr[t * 192 + r] * inv);
  }
  __syncthreads();

  // ============ scores = z_norm @ scratch_norm^T (64x64) ============
  {
    const unsigned char* sn = ws + WS_SN;
    #pragma unroll
    for (int i = 0; i < 2; ++i) {
      const int flat = wv * 2 + i, mtile = flat >> 2, ntile = flat & 3;
      v8f acc = zero8();
      const unsigned char* abase = (const unsigned char*)stage + (mtile * 16 + l16) * 256;
      const unsigned char* bbase = sn + (ntile * 16 + l16) * 256;
      #pragma unroll
      for (int k4 = 0; k4 < 4; ++k4) {
        FragU ua, ub;
        ua.q[0] = *(const v4u*)(abase + k4 * 64 + g * 32);
        ua.q[1] = *(const v4u*)(abase + k4 * 64 + g * 32 + 16);
        ub.q[0] = *(const v4u*)(bbase + k4 * 64 + g * 32);
        ub.q[1] = *(const v4u*)(bbase + k4 * 64 + g * 32 + 16);
        acc = wmma_bf16(acc, ua.v, ub.v);
      }
      #pragma unroll
      for (int j = 0; j < 8; ++j)
        zlr[(mtile * 16 + j + g * 8) * 192 + ntile * 16 + l16] = acc[j];
    }
  }
  __syncthreads();

  // ============ B2: softmax(+null), stats, mode_probs (bf16 in stage) ============
  if (tid < 64) {
    const int t = tid;
    float res = 0.f, mx = 0.f;
    for (int k = 0; k < 64; ++k) {
      const float sr = zlr[t * 192 + k];
      res += sr;
      const float sd = (sr - cmean[k]) / fmaxf(cstd[k], EPSF) - thr[k];
      zlr[t * 192 + k] = sd;
      mx = fmaxf(mx, sd);
    }
    float den = __expf(-mx);                               // null logit 0
    for (int k = 0; k < 64; ++k) den += __expf(zlr[t * 192 + k] - mx);
    const float invd = 1.0f / den;
    float ent = 0.f;
    for (int k = 0; k < 64; ++k) {
      const float p = __expf(zlr[t * 192 + k] - mx) * invd;
      zlr[t * 192 + k] = p;
      stage[t * 64 + k] = (__bf16)p;                       // mode_probs (A matrix)
      const float rc = fmaxf(p, EPSF);
      ent -= rc * __logf(rc);
    }
    const float pn = __expf(-mx) * invd;
    const float rcn = fmaxf(pn, EPSF);
    ent -= rcn * __logf(rcn);
    red[t] = ent; red[64 + t] = pn; red[128 + t] = res;
  }
  __syncthreads();
  if (tid < 64) {                                           // mode_usage[k]
    float s = 0.f;
    for (int t = 0; t < 64; ++t) s += zlr[t * 192 + tid];
    atomicAdd(&stats[tid], s * invN);
  } else if (tid == 64) {
    float s = 0.f; for (int t = 0; t < 64; ++t) s += red[t];
    atomicAdd(&stats[65], s * invN);                        // routing_entropy
  } else if (tid == 65) {
    float s = 0.f; for (int t = 0; t < 64; ++t) s += red[64 + t];
    atomicAdd(&stats[64], s * invN);                        // null_usage
  } else if (tid == 66) {
    float s = 0.f; for (int t = 0; t < 64; ++t) s += red[128 + t];
    atomicAdd(&stats[66], s * invN * (1.0f / 64.0f));       // mean_resonance
  }
  __syncthreads();

  // ============ B3: weighted_gate = mode_probs @ gate (64x64) ============
  {
    const unsigned char* gt = ws + WS_GATE;
    #pragma unroll
    for (int i = 0; i < 2; ++i) {
      const int flat = wv * 2 + i, mtile = flat >> 2, ntile = flat & 3;
      v8f acc = zero8();
      const unsigned char* abase = (const unsigned char*)stage + (mtile * 16 + l16) * 128;
      const unsigned char* bbase = gt + (ntile * 16 + l16) * 128;
      #pragma unroll
      for (int k2 = 0; k2 < 2; ++k2) {
        FragU ua, ub;
        ua.q[0] = *(const v4u*)(abase + k2 * 64 + g * 32);
        ua.q[1] = *(const v4u*)(abase + k2 * 64 + g * 32 + 16);
        ub.q[0] = *(const v4u*)(bbase + k2 * 64 + g * 32);
        ub.q[1] = *(const v4u*)(bbase + k2 * 64 + g * 32 + 16);
        acc = wmma_bf16(acc, ua.v, ub.v);
      }
      #pragma unroll
      for (int j = 0; j < 8; ++j)
        zlr[(mtile * 16 + j + g * 8) * 192 + ntile * 16 + l16] = acc[j];
    }
  }
  __syncthreads();

  // ============ B4: x = ALPHA * lr * wg (bf16 A matrix) ============
  {
    #pragma unroll
    for (int e = 0; e < 16; ++e) {
      const int idx = tid + e * 256;         // 0..4095 == t*64 + r
      const int t = idx >> 6, r = idx & 63;
      const float x = ALPHAF * zlr[t * 192 + 128 + r] * zlr[t * 192 + r];
      stage[4096 + idx] = (__bf16)x;
    }
    if (tid < 64) nrm[tid] = 0.f;
  }
  __syncthreads();

  // ============ Phase C: delta = x @ Wu^T, modulated = hid + delta ============
  {
    const unsigned char* wu = ws + WS_WU;
    const __bf16* smX = stage + 4096;
    const int mtile = wv >> 1;
    float dn[8];
    #pragma unroll
    for (int j = 0; j < 8; ++j) dn[j] = 0.f;
    const unsigned char* abase = (const unsigned char*)smX + (mtile * 16 + l16) * 128;
    FragU ua0, ua1;
    ua0.q[0] = *(const v4u*)(abase + g * 32);
    ua0.q[1] = *(const v4u*)(abase + g * 32 + 16);
    ua1.q[0] = *(const v4u*)(abase + 64 + g * 32);
    ua1.q[1] = *(const v4u*)(abase + 64 + g * 32 + 16);

    for (int hc = 0; hc < 16; ++hc) {
      const int h0 = hc * 128;
      #pragma unroll
      for (int i = 0; i < 4; ++i) {
        const int ntile = (wv & 1) * 4 + i;
        const int h = h0 + ntile * 16 + l16;
        const unsigned char* bbase = wu + (size_t)h * 128;
        v8f acc = zero8();
        FragU ub;
        ub.q[0] = *(const v4u*)(bbase + g * 32);
        ub.q[1] = *(const v4u*)(bbase + g * 32 + 16);
        acc = wmma_bf16(acc, ua0.v, ub.v);
        ub.q[0] = *(const v4u*)(bbase + 64 + g * 32);
        ub.q[1] = *(const v4u*)(bbase + 64 + g * 32 + 16);
        acc = wmma_bf16(acc, ua1.v, ub.v);
        #pragma unroll
        for (int j = 0; j < 8; ++j) {
          const int m = mtile * 16 + j + g * 8;
          const size_t gi = (size_t)(T0 + m) * Hh + h;
          const float d = acc[j];
          out[gi] = hid[gi] + d;
          dn[j] += d * d;
        }
      }
    }
    #pragma unroll
    for (int j = 0; j < 8; ++j)
      atomicAdd(&nrm[mtile * 16 + j + g * 8], dn[j]);
  }
  __syncthreads();
  if (tid < 64) atomicAdd(&stats[67], sqrtf(nrm[tid]) * invN);   // mean_shift_norm
}

// ---------------------------------------------------------------------------
// Precompute: gateT, scratch_norm, Wu (all bf16 in ws); zero stat accumulators
// ---------------------------------------------------------------------------
extern "C" __global__ __launch_bounds__(128)
void pcl_pre(const float* __restrict__ Wg, const float* __restrict__ interv,
             const float* __restrict__ scratch, const float* __restrict__ Wu,
             unsigned char* __restrict__ ws, float* __restrict__ stats)
{
  const int bid = blockIdx.x, tid = threadIdx.x;
  if (bid < 64) {
    if (tid < 64) {
      const int k = bid, r = tid;
      float s = 0.f;
      for (int h = 0; h < Hh; ++h) s += interv[(size_t)k * Hh + h] * Wg[(size_t)r * Hh + h];
      const float gv = 2.0f / (1.0f + __expf(-s)) - 1.0f;
      ((__bf16*)(ws + WS_GATE))[r * 64 + k] = (__bf16)gv;   // transposed: row r, contiguous k
    }
  } else if (bid == 64) {
    if (tid < 64) {
      const int k = tid;
      float ss = 0.f;
      for (int r = 0; r < 128; ++r) { const float v = scratch[k * 128 + r]; ss += v * v; }
      const float inv = 1.0f / fmaxf(sqrtf(ss), EPSF);
      __bf16* sn = (__bf16*)(ws + WS_SN);
      for (int r = 0; r < 128; ++r) sn[k * 128 + r] = (__bf16)(scratch[k * 128 + r] * inv);
    }
  } else if (bid == 65) {
    if (tid < 68) stats[tid] = 0.f;
  } else {                                                   // blocks 66..97: Wu -> bf16
    const int base = (bid - 66) * 4096;
    __bf16* wu = (__bf16*)(ws + WS_WU);
    for (int e = 0; e < 32; ++e) {
      const int idx = base + tid + e * 128;
      wu[idx] = (__bf16)Wu[idx];
    }
  }
}

extern "C" void kernel_launch(void* const* d_in, const int* in_sizes, int n_in,
                              void* d_out, int out_size, void* d_ws, size_t ws_size,
                              hipStream_t stream) {
  const float* hid = (const float*)d_in[0];
  const float* Wr  = (const float*)d_in[1];
  const float* Wd  = (const float*)d_in[2];
  const float* Wg  = (const float*)d_in[3];
  const float* Wu  = (const float*)d_in[4];
  const float* scr = (const float*)d_in[5];
  const float* itv = (const float*)d_in[6];
  const float* thr = (const float*)d_in[7];
  const float* cme = (const float*)d_in[8];
  const float* cst = (const float*)d_in[9];
  float* out = (float*)d_out;
  unsigned char* ws = (unsigned char*)d_ws;
  float* stats = out + (size_t)NTOK * Hh;

  pcl_pre<<<dim3(98), dim3(128), 0, stream>>>(Wg, itv, scr, Wu, ws, stats);
  pcl_main<<<dim3(NTOK / MT), dim3(256), 0, stream>>>(hid, Wr, Wd, thr, cme, cst, ws, out);
}
